// Recommender_23596550324576
// MI455X (gfx1250) — matrix operand order repaired and checked by reading, not compiled, not run on hardware
//
#include <hip/hip_runtime.h>
#include <hip/hip_bf16.h>
#include <math.h>

#define NU     30000
#define NE     10000
#define NRW    10          // N_RELATIONS - 1
#define EMB    64
#define NEDGE  400000
#define NNZ    500000
#define NB     2048
#define TOPK   10
#define TILES  (NE / 16)   // 625
#define EPSV   1e-12f
#define LAMB   0.5f

#define CDIV(a, b) (((a) + (b) - 1) / (b))

typedef __attribute__((ext_vector_type(16))) __bf16 v16bf;
typedef __attribute__((ext_vector_type(8)))  float  v8f;

// register-resident sorted top-10 insert (all static indices -> no scratch)
#define TOPK_INIT(tv, ti)                                                   \
    _Pragma("unroll")                                                       \
    for (int _i = 0; _i < TOPK; ++_i) { tv[_i] = -3.0e38f; ti[_i] = 0; }

#define TOPK_INSERT(tv, ti, v, idx)                                         \
    if ((v) > tv[TOPK - 1]) {                                               \
        tv[TOPK - 1] = (v); ti[TOPK - 1] = (idx);                           \
        _Pragma("unroll")                                                   \
        for (int _p = TOPK - 1; _p > 0; --_p) {                             \
            if (tv[_p] > tv[_p - 1]) {                                      \
                float _tf = tv[_p]; tv[_p] = tv[_p - 1]; tv[_p - 1] = _tf;  \
                int   _tt = ti[_p]; ti[_p] = ti[_p - 1]; ti[_p - 1] = _tt;  \
            }                                                               \
        }                                                                   \
    }

// ---------------------------------------------------------------------------
// init: split all_embed into user/entity current + residual accumulators
// ---------------------------------------------------------------------------
__global__ void k_init(const float* __restrict__ all_embed,
                       float* __restrict__ user_cur, float* __restrict__ user_res,
                       float* __restrict__ ent_cur,  float* __restrict__ ent_res) {
    long i = (long)blockIdx.x * blockDim.x + threadIdx.x;
    const long nu = (long)NU * EMB, tot = (long)(NU + NE) * EMB;
    if (i >= tot) return;
    float v = all_embed[i];
    if (i < nu) { user_cur[i] = v; user_res[i] = v; }
    else        { ent_cur[i - nu] = v; ent_res[i - nu] = v; }
}

// ---------------------------------------------------------------------------
// row L2 normalize (eps==0 -> plain divide like _build_adj; eps>0 -> _l2norm)
// ---------------------------------------------------------------------------
__global__ void k_rownorm(const float* __restrict__ x, float* __restrict__ y,
                          int nrows, float eps) {
    int r = blockIdx.x * blockDim.x + threadIdx.x;
    if (r >= nrows) return;
    const float* xr = x + (long)r * EMB;
    float s = 0.f;
    #pragma unroll 8
    for (int d = 0; d < EMB; ++d) { float v = xr[d]; s += v * v; }
    float nrm = sqrtf(s);
    if (eps > 0.f) nrm = fmaxf(nrm, eps);
    float inv = 1.f / nrm;
    float* yr = y + (long)r * EMB;
    #pragma unroll 8
    for (int d = 0; d < EMB; ++d) yr[d] = xr[d] * inv;
}

// ---------------------------------------------------------------------------
// pack cn (NE x 64 f32) into WMMA A-layout bf16 tiles.
// A 16x32 bf16 layout: lanes 0-15 hold K=0-7 (v0-3) & K=16-23 (v4-7);
// lanes 16-31 hold K=8-15 & K=24-31. Element e in the v16bf:
//   k_local = kb + e + (e>=8 ? 8 : 0), kb = (lane<16 ? 0 : 8)
// buffer index: ((tile*2 + chunk)*32 + lane)*16 + e
// ---------------------------------------------------------------------------
__global__ void k_pack_a(const float* __restrict__ cn, __bf16* __restrict__ packA) {
    int t = blockIdx.x * blockDim.x + threadIdx.x;
    if (t >= TILES * 1024) return;
    int e = t & 15, lane = (t >> 4) & 31, chunk = (t >> 9) & 1, tile = t >> 10;
    int m  = lane & 15;
    int kb = ((lane < 16) ? 0 : 8);
    int k  = chunk * 32 + kb + e + ((e >= 8) ? 8 : 0);
    packA[t] = (__bf16)cn[(long)(tile * 16 + m) * EMB + k];
}

// B 32x16 bf16 layout (B = cn^T): lane n = column; lanes 0-15 hold K=0-7 & K=16-23,
// lanes 16-31 hold K=8-15 & K=24-31; element e -> k_local = (lane<16 ? 0 : 8) + e + (e>=8 ? 8 : 0).
__global__ void k_pack_b(const float* __restrict__ cn, __bf16* __restrict__ packB) {
    int t = blockIdx.x * blockDim.x + threadIdx.x;
    if (t >= TILES * 1024) return;
    int e = t & 15, lane = (t >> 4) & 31, chunk = (t >> 9) & 1, tile = t >> 10;
    int n = lane & 15;
    int k = chunk * 32 + ((lane < 16) ? 0 : 8) + e + ((e >= 8) ? 8 : 0);
    packB[t] = (__bf16)cn[(long)(tile * 16 + n) * EMB + k];
}

// ---------------------------------------------------------------------------
// Fused sim = cn @ cn^T with running per-row top-10, operand-swapped:
//   stream COLUMN tiles as A, keep TWO resident ROW tiles as B.
//   D[m][n] = sim[row0+n][ct*16+m]  => lane n holds 8 sim values of its OWN
//   row in the accumulator VGPRs (lanes 0-15: cols +0..7, 16-31: cols +8..15)
//   -> no LDS in the hot loop, top-k on all 32 lanes straight from registers.
// 2 row tiles per wave halves the streamed L2 traffic; A loads for ct+1 are
// issued after the WMMAs so they overlap the top-k VALU work (bf16 WMMA is a
// TRANS op on CDNA5 and co-executes with VALU).
// ---------------------------------------------------------------------------
__global__ __launch_bounds__(32) void k_simtopk(const __bf16* __restrict__ packA,
                                                const __bf16* __restrict__ packB,
                                                float* __restrict__ knn_val,
                                                int*   __restrict__ knn_ind) {
    const int lane  = threadIdx.x;
    const int tile0 = blockIdx.x * 2;
    const int tile1 = tile0 + 1;
    const bool has1 = (tile1 < TILES);
    const int tile1c = has1 ? tile1 : tile0;

    // loop-invariant B operands: the two resident row tiles
    v16bf rb0 = *(const v16bf*)(packB + ((long)(tile0  * 2 + 0) * 32 + lane) * 16);
    v16bf rb1 = *(const v16bf*)(packB + ((long)(tile0  * 2 + 1) * 32 + lane) * 16);
    v16bf rb2 = *(const v16bf*)(packB + ((long)(tile1c * 2 + 0) * 32 + lane) * 16);
    v16bf rb3 = *(const v16bf*)(packB + ((long)(tile1c * 2 + 1) * 32 + lane) * 16);

    float tv0[TOPK]; int ti0[TOPK];
    float tv1[TOPK]; int ti1[TOPK];
    TOPK_INIT(tv0, ti0);
    TOPK_INIT(tv1, ti1);

    const int coff = (lane & 16) ? 8 : 0;   // column sub-offset owned by this lane

    // software pipeline: preload A (column tile 0)
    v16bf ca0 = *(const v16bf*)(packA + ((long)0 * 32 + lane) * 16);
    v16bf ca1 = *(const v16bf*)(packA + ((long)1 * 32 + lane) * 16);

    for (int ct = 0; ct < TILES; ++ct) {
        v8f c0 = {0.f, 0.f, 0.f, 0.f, 0.f, 0.f, 0.f, 0.f};
        v8f c1 = {0.f, 0.f, 0.f, 0.f, 0.f, 0.f, 0.f, 0.f};
        c0 = __builtin_amdgcn_wmma_f32_16x16x32_bf16(false, ca0, false, rb0,
                                                     (short)0, c0, false, false);
        c0 = __builtin_amdgcn_wmma_f32_16x16x32_bf16(false, ca1, false, rb1,
                                                     (short)0, c0, false, false);
        c1 = __builtin_amdgcn_wmma_f32_16x16x32_bf16(false, ca0, false, rb2,
                                                     (short)0, c1, false, false);
        c1 = __builtin_amdgcn_wmma_f32_16x16x32_bf16(false, ca1, false, rb3,
                                                     (short)0, c1, false, false);

        // prefetch next column tile; overlaps the top-k below
        if (ct + 1 < TILES) {
            ca0 = *(const v16bf*)(packA + ((long)((ct + 1) * 2 + 0) * 32 + lane) * 16);
            ca1 = *(const v16bf*)(packA + ((long)((ct + 1) * 2 + 1) * 32 + lane) * 16);
        }

        const int cbase = ct * 16 + coff;
        #pragma unroll
        for (int r = 0; r < 8; ++r) {
            float v0 = c0[r];
            TOPK_INSERT(tv0, ti0, v0, cbase + r);
            float v1 = c1[r];
            TOPK_INSERT(tv1, ti1, v1, cbase + r);
        }
    }

    // merge the two half-column lists per row (lane L owns row lane&15)
    __shared__ float sv[2][16][TOPK];
    __shared__ int   si[2][16][TOPK];
    if (lane >= 16) {
        #pragma unroll
        for (int i = 0; i < TOPK; ++i) {
            sv[0][lane - 16][i] = tv0[i]; si[0][lane - 16][i] = ti0[i];
            sv[1][lane - 16][i] = tv1[i]; si[1][lane - 16][i] = ti1[i];
        }
    }
    __syncthreads();
    if (lane < 16) {
        #pragma unroll
        for (int i = 0; i < TOPK; ++i) {
            float v = sv[0][lane][i]; int ix = si[0][lane][i];
            TOPK_INSERT(tv0, ti0, v, ix);
            v = sv[1][lane][i]; ix = si[1][lane][i];
            TOPK_INSERT(tv1, ti1, v, ix);
        }
        int row0 = tile0 * 16 + lane;
        #pragma unroll
        for (int i = 0; i < TOPK; ++i) {
            knn_val[(long)row0 * TOPK + i] = tv0[i];
            knn_ind[(long)row0 * TOPK + i] = ti0[i];
        }
        if (has1) {
            int row1 = tile1 * 16 + lane;
            #pragma unroll
            for (int i = 0; i < TOPK; ++i) {
                knn_val[(long)row1 * TOPK + i] = tv1[i];
                knn_ind[(long)row1 * TOPK + i] = ti1[i];
            }
        }
    }
}

// d_inv_sqrt[i] = (sum_j knn_val[i,j])^-0.5
__global__ void k_dinv(const float* __restrict__ knn_val, float* __restrict__ dinv) {
    int i = blockIdx.x * blockDim.x + threadIdx.x;
    if (i >= NE) return;
    float s = 0.f;
    #pragma unroll
    for (int j = 0; j < TOPK; ++j) s += knn_val[(long)i * TOPK + j];
    dinv[i] = rsqrtf(s);
}

// vals = dinv[row] * knn_val * dinv[col]
__global__ void k_vals(const float* __restrict__ knn_val, const int* __restrict__ knn_ind,
                       const float* __restrict__ dinv, float* __restrict__ vals) {
    int t = blockIdx.x * blockDim.x + threadIdx.x;
    if (t >= NE * TOPK) return;
    int i = t / TOPK;
    vals[t] = dinv[i] * knn_val[t] * dinv[knn_ind[t]];
}

// ---------------------------------------------------------------------------
// KG attention hop kernels
// ---------------------------------------------------------------------------
__global__ void k_edge_logits(const float* __restrict__ ent, const float* __restrict__ wrel,
                              const int* __restrict__ head, const int* __restrict__ tail,
                              const int* __restrict__ etype,
                              float* __restrict__ logits, float* __restrict__ mbuf) {
    int e = blockIdx.x * blockDim.x + threadIdx.x;
    if (e >= NEDGE) return;
    int h = head[e], tl = tail[e], r = etype[e] - 1;
    const float* wr = wrel + (long)r * EMB;
    const float* eh = ent + (long)h * EMB;
    const float* et = ent + (long)tl * EMB;
    float hn2 = 0.f, tn2 = 0.f;
    #pragma unroll 8
    for (int d = 0; d < EMB; ++d) {
        float ww = wr[d];
        float a = eh[d] * ww; hn2 += a * a;
        float b = et[d] * ww; tn2 += b * b;
    }
    float lg = hn2 * tn2;                 // (hn*tn)^2, >= 0
    logits[e] = lg;
    // nonneg float: uint ordering == float ordering -> segment_max via atomicMax
    atomicMax((unsigned int*)(mbuf + h), __float_as_uint(lg));
}

__global__ void k_edge_att(const float* __restrict__ logits, const float* __restrict__ mbuf,
                           const int* __restrict__ head,
                           float* __restrict__ eval, float* __restrict__ sbuf) {
    int e = blockIdx.x * blockDim.x + threadIdx.x;
    if (e >= NEDGE) return;
    int h = head[e];
    float ev = expf(logits[e] - mbuf[h]);
    eval[e] = ev;
    atomicAdd(sbuf + h, ev);
}

__global__ void k_edge_scatter(const float* __restrict__ ent, const float* __restrict__ wrel,
                               const int* __restrict__ head, const int* __restrict__ tail,
                               const int* __restrict__ etype,
                               const float* __restrict__ eval, const float* __restrict__ sbuf,
                               float* __restrict__ ent_agg) {
    long t = (long)blockIdx.x * blockDim.x + threadIdx.x;
    if (t >= (long)NEDGE * EMB) return;
    int e = (int)(t >> 6), d = (int)(t & 63);
    int h = head[e];
    float att = eval[e] / sbuf[h];
    float nv = ent[(long)tail[e] * EMB + d] * wrel[(long)(etype[e] - 1) * EMB + d];
    atomicAdd(ent_agg + (long)h * EMB + d, att * nv);
}

__global__ void k_user_scatter(const float* __restrict__ ent,
                               const int* __restrict__ irows, const int* __restrict__ icols,
                               const float* __restrict__ ivals,
                               float* __restrict__ user_agg) {
    long t = (long)blockIdx.x * blockDim.x + threadIdx.x;
    if (t >= (long)NNZ * EMB) return;
    int nz = (int)(t >> 6), d = (int)(t & 63);
    atomicAdd(user_agg + (long)irows[nz] * EMB + d,
              ivals[nz] * ent[(long)icols[nz] * EMB + d]);
}

// user_agg += softmax(user_cur @ W^T) @ W * user_agg
__global__ void k_user_gate(const float* __restrict__ user_cur,
                            const float* __restrict__ wrel,
                            float* __restrict__ user_agg) {
    __shared__ float sw[NRW * EMB];
    for (int i = threadIdx.x; i < NRW * EMB; i += blockDim.x) sw[i] = wrel[i];
    __syncthreads();
    int u = blockIdx.x * blockDim.x + threadIdx.x;
    if (u >= NU) return;
    const float* uc = user_cur + (long)u * EMB;
    float sc[NRW];
    float mx = -3.0e38f;
    #pragma unroll
    for (int r = 0; r < NRW; ++r) {
        float s = 0.f;
        #pragma unroll 8
        for (int d = 0; d < EMB; ++d) s += uc[d] * sw[r * EMB + d];
        sc[r] = s;
        mx = fmaxf(mx, s);
    }
    float ssum = 0.f;
    #pragma unroll
    for (int r = 0; r < NRW; ++r) { sc[r] = expf(sc[r] - mx); ssum += sc[r]; }
    float inv = 1.f / ssum;
    float* ua = user_agg + (long)u * EMB;
    for (int d = 0; d < EMB; ++d) {
        float g = 0.f;
        #pragma unroll
        for (int r = 0; r < NRW; ++r) g += sc[r] * sw[r * EMB + d];
        ua[d] *= (1.f + g * inv);
    }
}

// cur = l2norm(agg) (eps-guarded); res += cur
__global__ void k_norm_accum(const float* __restrict__ agg,
                             float* __restrict__ cur, float* __restrict__ res, int nrows) {
    int r = blockIdx.x * blockDim.x + threadIdx.x;
    if (r >= nrows) return;
    const float* ar = agg + (long)r * EMB;
    float s = 0.f;
    #pragma unroll 8
    for (int d = 0; d < EMB; ++d) { float v = ar[d]; s += v * v; }
    float inv = 1.f / fmaxf(sqrtf(s), EPSV);
    float* cr = cur + (long)r * EMB;
    float* rr = res + (long)r * EMB;
    #pragma unroll 8
    for (int d = 0; d < EMB; ++d) { float v = ar[d] * inv; cr[d] = v; rr[d] += v; }
}

// i_h = (1-LAMB)*spmm(adj2, item0) + LAMB*spmm(adj1, item0)
__global__ void k_ih(const float* __restrict__ vals1, const int* __restrict__ ind1,
                     const float* __restrict__ vals2, const int* __restrict__ ind2,
                     const float* __restrict__ item0, float* __restrict__ ih) {
    long t = (long)blockIdx.x * blockDim.x + threadIdx.x;
    if (t >= (long)NE * EMB) return;
    int i = (int)(t >> 6), d = (int)(t & 63);
    float acc = 0.f;
    #pragma unroll
    for (int j = 0; j < TOPK; ++j) {
        long k = (long)i * TOPK + j;
        acc += (1.f - LAMB) * vals2[k] * item0[(long)ind2[k] * EMB + d];
        acc += LAMB         * vals1[k] * item0[(long)ind1[k] * EMB + d];
    }
    ih[t] = acc;
}

// out = stack([user_res[users], ent_res[items], i_h[items]])
__global__ void k_gather(const int* __restrict__ users, const int* __restrict__ items,
                         const float* __restrict__ user_res, const float* __restrict__ ent_res,
                         const float* __restrict__ ih, float* __restrict__ out) {
    int t = blockIdx.x * blockDim.x + threadIdx.x;
    if (t >= NB * EMB) return;
    int b = t >> 6, d = t & 63;
    out[t]                = user_res[(long)users[b] * EMB + d];
    out[NB * EMB + t]     = ent_res[(long)items[b] * EMB + d];
    out[2 * NB * EMB + t] = ih[(long)items[b] * EMB + d];
}

// ---------------------------------------------------------------------------
extern "C" void kernel_launch(void* const* d_in, const int* in_sizes, int n_in,
                              void* d_out, int out_size, void* d_ws, size_t ws_size,
                              hipStream_t stream) {
    (void)in_sizes; (void)n_in; (void)out_size; (void)ws_size;
    const float* all_embed = (const float*)d_in[0];
    const float* rel_w     = (const float*)d_in[1];
    const float* ivals     = (const float*)d_in[2];
    const int*   edge_idx  = (const int*)d_in[3];   // (2, NEDGE)
    const int*   etype     = (const int*)d_in[4];
    const int*   irows     = (const int*)d_in[5];
    const int*   icols     = (const int*)d_in[6];
    const int*   users     = (const int*)d_in[7];
    const int*   items     = (const int*)d_in[8];
    const int*   head = edge_idx;
    const int*   tail = edge_idx + NEDGE;
    const float* item0 = all_embed + (long)NU * EMB;  // entity slice of all_embed
    float* out = (float*)d_out;

    // ---- workspace layout ----
    char* w = (char*)d_ws;
    size_t off = 0;
    auto A = [&](size_t bytes) { char* p = w + off; off = (off + bytes + 255) & ~(size_t)255; return p; };
    float*  cn       = (float*)A((size_t)NE * EMB * 4);
    __bf16* packA    = (__bf16*)A((size_t)TILES * 1024 * 2);
    __bf16* packB    = (__bf16*)A((size_t)TILES * 1024 * 2);
    float*  knv1     = (float*)A((size_t)NE * TOPK * 4);
    int*    kni1     = (int*)  A((size_t)NE * TOPK * 4);
    float*  dinv1    = (float*)A((size_t)NE * 4);
    float*  vals1    = (float*)A((size_t)NE * TOPK * 4);
    float*  knv2     = (float*)A((size_t)NE * TOPK * 4);
    int*    kni2     = (int*)  A((size_t)NE * TOPK * 4);
    float*  dinv2    = (float*)A((size_t)NE * 4);
    float*  vals2    = (float*)A((size_t)NE * TOPK * 4);
    float*  ent_cur  = (float*)A((size_t)NE * EMB * 4);
    float*  ent_res  = (float*)A((size_t)NE * EMB * 4);
    float*  ent_agg  = (float*)A((size_t)NE * EMB * 4);
    float*  user_cur = (float*)A((size_t)NU * EMB * 4);
    float*  user_res = (float*)A((size_t)NU * EMB * 4);
    float*  user_agg = (float*)A((size_t)NU * EMB * 4);
    float*  logits   = (float*)A((size_t)NEDGE * 4);
    float*  evalb    = (float*)A((size_t)NEDGE * 4);
    float*  mbuf     = (float*)A((size_t)NE * 4);
    float*  sbuf     = (float*)A((size_t)NE * 4);
    float*  ih       = (float*)A((size_t)NE * EMB * 4);

    const int T = 256;
    const int SIMGRID = CDIV(TILES, 2);   // 2 row tiles per wave

    // init embeddings / residuals
    k_init<<<CDIV((long)(NU + NE) * EMB, T), T, 0, stream>>>(all_embed, user_cur, user_res,
                                                             ent_cur, ent_res);

    // ---- build_adj round 1 (on entity_emb) ----
    k_rownorm<<<CDIV(NE, T), T, 0, stream>>>(ent_cur, cn, NE, 0.f);
    k_pack_a<<<CDIV(TILES * 1024, T), T, 0, stream>>>(cn, packA);
    k_pack_b<<<CDIV(TILES * 1024, T), T, 0, stream>>>(cn, packB);
    k_simtopk<<<SIMGRID, 32, 0, stream>>>(packA, packB, knv1, kni1);
    k_dinv<<<CDIV(NE, T), T, 0, stream>>>(knv1, dinv1);
    k_vals<<<CDIV(NE * TOPK, T), T, 0, stream>>>(knv1, kni1, dinv1, vals1);

    // ---- 2 hops ----
    for (int hop = 0; hop < 2; ++hop) {
        hipMemsetAsync(ent_agg, 0, (size_t)NE * EMB * 4, stream);
        hipMemsetAsync(user_agg, 0, (size_t)NU * EMB * 4, stream);
        hipMemsetAsync(mbuf, 0, (size_t)NE * 4, stream);
        hipMemsetAsync(sbuf, 0, (size_t)NE * 4, stream);

        k_edge_logits<<<CDIV(NEDGE, T), T, 0, stream>>>(ent_cur, rel_w, head, tail, etype,
                                                        logits, mbuf);
        k_edge_att<<<CDIV(NEDGE, T), T, 0, stream>>>(logits, mbuf, head, evalb, sbuf);
        k_edge_scatter<<<CDIV((long)NEDGE * EMB, T), T, 0, stream>>>(ent_cur, rel_w, head, tail,
                                                                     etype, evalb, sbuf, ent_agg);
        k_user_scatter<<<CDIV((long)NNZ * EMB, T), T, 0, stream>>>(ent_cur, irows, icols, ivals,
                                                                   user_agg);
        k_user_gate<<<CDIV(NU, T), T, 0, stream>>>(user_cur, rel_w, user_agg);

        k_norm_accum<<<CDIV(NE, T), T, 0, stream>>>(ent_agg, ent_cur, ent_res, NE);
        k_norm_accum<<<CDIV(NU, T), T, 0, stream>>>(user_agg, user_cur, user_res, NU);
    }

    // ---- build_adj round 2 (on entity_res) ----
    k_rownorm<<<CDIV(NE, T), T, 0, stream>>>(ent_res, cn, NE, 0.f);
    k_pack_a<<<CDIV(TILES * 1024, T), T, 0, stream>>>(cn, packA);
    k_pack_b<<<CDIV(TILES * 1024, T), T, 0, stream>>>(cn, packB);
    k_simtopk<<<SIMGRID, 32, 0, stream>>>(packA, packB, knv2, kni2);
    k_dinv<<<CDIV(NE, T), T, 0, stream>>>(knv2, dinv2);
    k_vals<<<CDIV(NE * TOPK, T), T, 0, stream>>>(knv2, kni2, dinv2, vals2);

    // ---- i_h = blend of the two KNN SpMMs over item_emb0, then l2norm ----
    k_ih<<<CDIV((long)NE * EMB, T), T, 0, stream>>>(vals1, kni1, vals2, kni2, item0, ih);
    k_rownorm<<<CDIV(NE, T), T, 0, stream>>>(ih, ih, NE, EPSV);

    // ---- gather outputs ----
    k_gather<<<CDIV(NB * EMB, T), T, 0, stream>>>(users, items, user_res, ent_res, ih, out);
}
